// QuestionSpecificMLP_27290222199412
// MI455X (gfx1250) — matrix operand order repaired and checked by reading, not compiled, not run on hardware
//
#include <hip/hip_runtime.h>
#include <stdint.h>

typedef __attribute__((ext_vector_type(4)))  float    v4f;
typedef __attribute__((ext_vector_type(8)))  float    v8f;
typedef __attribute__((ext_vector_type(16))) _Float16 v16h;

#define D_IN 385
#define D_H  192
#define NQ   1000
#define NC   16
#define LDS_STRIDE 196          // 192 + 4 pad -> conflict-free transposed reads
#define WAVES_PER_BLOCK 4
#define ROWS_PER_WAVE 16

// K padded to multiples of 32 for the f16 WMMA path
#define NCH1 13                 // ceil(385/32) -> 416
#define NCH2 6                  // 192/32
// fragment-major pack: per (chunk,tile): 32 lanes * 8 u32 = 256 u32 (1 KB)
#define PACK_BLK 256
#define W1_PACK_N (NCH1 * 12 * PACK_BLK)   // 39936 u32
#define W2_PACK_N (NCH2 * 12 * PACK_BLK)   // 18432 u32

// ---------------------------------------------------------------------------
// Pre-pack: W[K x 192] f32 -> f16 hi/lo B-fragments in contiguous-k order.
// B VGPR j of lane (h,n) holds rows k = 32*kc + 16*h + 2*j (+0 lo16, +1 hi16)
// at column 16*t + n.  Output index = ((kc*12 + t)*32 + lane)*8 + j.
// Rows >= K are zero (pads 385 -> 416).
// ---------------------------------------------------------------------------
__global__ void pack_w_f16_kernel(const float* __restrict__ W, int K, int total,
                                  uint32_t* __restrict__ outH,
                                  uint32_t* __restrict__ outL)
{
    int gid = blockIdx.x * blockDim.x + threadIdx.x;
    if (gid >= total) return;
    const int j    = gid & 7;
    const int lane = (gid >> 3) & 31;
    const int rest = gid >> 8;           // kc*12 + t
    const int t    = rest % 12;
    const int kc   = rest / 12;
    const int h    = lane >> 4;
    const int n    = lane & 15;
    const int k    = kc * 32 + 16 * h + 2 * j;
    const int col  = 16 * t + n;

    float w0 = (k     < K) ? W[(size_t)k       * D_H + col] : 0.0f;
    float w1 = (k + 1 < K) ? W[(size_t)(k + 1) * D_H + col] : 0.0f;
    _Float16 h0 = (_Float16)w0;
    _Float16 h1 = (_Float16)w1;
    _Float16 l0 = (_Float16)(w0 - (float)h0);
    _Float16 l1 = (_Float16)(w1 - (float)h1);

    union { uint32_t u; _Float16 f[2]; } ph, pl;
    ph.f[0] = h0; ph.f[1] = h1;
    pl.f[0] = l0; pl.f[1] = l1;
    outH[gid] = ph.u;
    outL[gid] = pl.u;
}

// Two-tile interleaved split-precision accumulation:
// separates each dependent acc chain with an independent WMMA from the
// neighboring tile so the matrix pipe never waits on TRANS32_DEP.
#define WMMA3X2(accA, accB, ah, al, bh0, bl0, bh1, bl1)                         \
    do {                                                                        \
        (accA) = __builtin_amdgcn_wmma_f32_16x16x32_f16(                        \
            false, (ah), false, (bh0), (short)0, (accA), false, false);         \
        (accB) = __builtin_amdgcn_wmma_f32_16x16x32_f16(                        \
            false, (ah), false, (bh1), (short)0, (accB), false, false);         \
        (accA) = __builtin_amdgcn_wmma_f32_16x16x32_f16(                        \
            false, (ah), false, (bl0), (short)0, (accA), false, false);         \
        (accB) = __builtin_amdgcn_wmma_f32_16x16x32_f16(                        \
            false, (ah), false, (bl1), (short)0, (accB), false, false);         \
        (accA) = __builtin_amdgcn_wmma_f32_16x16x32_f16(                        \
            false, (al), false, (bh0), (short)0, (accA), false, false);         \
        (accB) = __builtin_amdgcn_wmma_f32_16x16x32_f16(                        \
            false, (al), false, (bh1), (short)0, (accB), false, false);         \
    } while (0)

// ---------------------------------------------------------------------------
// Fused main kernel. One wave32 owns 16 samples end-to-end.
// GEMM stages: split-precision f16 (a=ah+al, b=bh+bl; D += ah*bh + ah*bl + al*bh)
// on V_WMMA_F32_16X16X32_F16  -> ~22-bit effective mantissa, f32 accumulate.
// ---------------------------------------------------------------------------
__global__ __launch_bounds__(128) void qmlp_fused_kernel(
    const float*    __restrict__ x,
    const int*      __restrict__ qid,
    const int*      __restrict__ isc,
    const uint32_t* __restrict__ w1h, const uint32_t* __restrict__ w1l,
    const float*    __restrict__ b1,
    const uint32_t* __restrict__ w2h, const uint32_t* __restrict__ w2l,
    const float*    __restrict__ b2,
    const float*    __restrict__ Wh,
    const float*    __restrict__ bh,
    float*          __restrict__ out)
{
    __shared__ __align__(16) float smem[WAVES_PER_BLOCK][ROWS_PER_WAVE * LDS_STRIDE];

    const int tid  = threadIdx.x;
    const int wid  = tid >> 5;
    const int lane = tid & 31;
    const int m    = lane & 15;   // A/C tile row == B/C tile col for this lane
    const int h    = lane >> 4;   // lane half selects K-subrange per 16-bit layout
    const int row0 = (blockIdx.x * WAVES_PER_BLOCK + wid) * ROWS_PER_WAVE;
    const int lanebase = lane * 8;

    float* hbuf = &smem[wid][0];

    // ---------------- Stage 1: H[16x192] = relu(X[16x385] @ W1 + b1) --------
    v8f acc[12] = {};
    const float* xrow = x + (size_t)(row0 + m) * D_IN;

    #pragma unroll 1
    for (int kc = 0; kc < 12; ++kc) {
        // A fragment: two contiguous 8-float runs, split into f16 hi/lo
        v16h ah, al;
        const float* p = xrow + 32 * kc + 16 * h;
        #pragma unroll
        for (int e = 0; e < 8; ++e) {
            float v0 = p[e];
            float v1 = p[8 + e];
            _Float16 h0 = (_Float16)v0;
            _Float16 h1 = (_Float16)v1;
            ah[e]     = h0;
            ah[8 + e] = h1;
            al[e]     = (_Float16)(v0 - (float)h0);
            al[8 + e] = (_Float16)(v1 - (float)h1);
        }
        const uint32_t* bH = w1h + (size_t)(kc * 12) * PACK_BLK + lanebase;
        const uint32_t* bL = w1l + (size_t)(kc * 12) * PACK_BLK + lanebase;
        #pragma unroll
        for (int t = 0; t < 12; t += 2) {
            v16h bh0 = *(const v16h*)(bH + t * PACK_BLK);
            v16h bl0 = *(const v16h*)(bL + t * PACK_BLK);
            v16h bh1 = *(const v16h*)(bH + (t + 1) * PACK_BLK);
            v16h bl1 = *(const v16h*)(bL + (t + 1) * PACK_BLK);
            WMMA3X2(acc[t], acc[t + 1], ah, al, bh0, bl0, bh1, bl1);
        }
    }
    {   // remainder chunk kc=12: only k = 384 is live (pack already zero-padded)
        v16h ah, al;
        #pragma unroll
        for (int e = 0; e < 16; ++e) { ah[e] = (_Float16)0.0f; al[e] = (_Float16)0.0f; }
        float v0 = (h == 0) ? xrow[384] : 0.0f;
        _Float16 h0 = (_Float16)v0;
        ah[0] = h0;
        al[0] = (_Float16)(v0 - (float)h0);
        const uint32_t* bH = w1h + (size_t)(12 * 12) * PACK_BLK + lanebase;
        const uint32_t* bL = w1l + (size_t)(12 * 12) * PACK_BLK + lanebase;
        #pragma unroll
        for (int t = 0; t < 12; t += 2) {
            v16h bh0 = *(const v16h*)(bH + t * PACK_BLK);
            v16h bl0 = *(const v16h*)(bL + t * PACK_BLK);
            v16h bh1 = *(const v16h*)(bH + (t + 1) * PACK_BLK);
            v16h bl1 = *(const v16h*)(bL + (t + 1) * PACK_BLK);
            WMMA3X2(acc[t], acc[t + 1], ah, al, bh0, bl0, bh1, bl1);
        }
    }
    // bias + relu, C/D layout -> row-major f32 LDS [16][LDS_STRIDE]
    #pragma unroll
    for (int t = 0; t < 12; ++t) {
        const float bias = b1[16 * t + m];
        #pragma unroll
        for (int r = 0; r < 8; ++r) {
            float v = acc[t][r] + bias;
            hbuf[(r + 8 * h) * LDS_STRIDE + 16 * t + m] = v > 0.0f ? v : 0.0f;
        }
    }
    __syncthreads();

    // ---------------- Stage 2: F[16x192] = relu(H @ W2 + b2) ----------------
    v8f acc2[12] = {};
    #pragma unroll 1
    for (int kc = 0; kc < NCH2; ++kc) {
        const float* p = hbuf + m * LDS_STRIDE + 32 * kc + 16 * h;   // 16B aligned
        v4f f0 = *(const v4f*)(p);
        v4f f1 = *(const v4f*)(p + 4);
        v4f f2 = *(const v4f*)(p + 8);
        v4f f3 = *(const v4f*)(p + 12);
        float r1[8] = { f0.x, f0.y, f0.z, f0.w, f1.x, f1.y, f1.z, f1.w };
        float r2[8] = { f2.x, f2.y, f2.z, f2.w, f3.x, f3.y, f3.z, f3.w };
        v16h ah, al;
        #pragma unroll
        for (int e = 0; e < 8; ++e) {
            _Float16 h0 = (_Float16)r1[e];
            _Float16 h1 = (_Float16)r2[e];
            ah[e]     = h0;
            ah[8 + e] = h1;
            al[e]     = (_Float16)(r1[e] - (float)h0);
            al[8 + e] = (_Float16)(r2[e] - (float)h1);
        }
        const uint32_t* bH = w2h + (size_t)(kc * 12) * PACK_BLK + lanebase;
        const uint32_t* bL = w2l + (size_t)(kc * 12) * PACK_BLK + lanebase;
        #pragma unroll
        for (int t = 0; t < 12; t += 2) {
            v16h bh0 = *(const v16h*)(bH + t * PACK_BLK);
            v16h bl0 = *(const v16h*)(bL + t * PACK_BLK);
            v16h bh1 = *(const v16h*)(bH + (t + 1) * PACK_BLK);
            v16h bl1 = *(const v16h*)(bL + (t + 1) * PACK_BLK);
            WMMA3X2(acc2[t], acc2[t + 1], ah, al, bh0, bl0, bh1, bl1);
        }
    }
    __syncthreads();   // drain H reads before overwriting buffer with F
    #pragma unroll
    for (int t = 0; t < 12; ++t) {
        const float bias = b2[16 * t + m];
        #pragma unroll
        for (int r = 0; r < 8; ++r) {
            float v = acc2[t][r] + bias;
            hbuf[(r + 8 * h) * LDS_STRIDE + 16 * t + m] = v > 0.0f ? v : 0.0f;
        }
    }
    __syncthreads();

    // ------------- Stage 3: routed heads, per-sample [16x192] mat-vec -------
    // lane -> sample s = m, classes [h*8, h*8+8). Head bank (24.6 MB) is L2-resident.
    const float* feat = &hbuf[m * LDS_STRIDE];
    const int row  = row0 + m;
    const int hidx = qid[row] + isc[row] * NQ;
    const float* wbase = Wh + (size_t)hidx * (NC * D_H) + (size_t)(h * 8) * D_H;

    float acc3[8];
    #pragma unroll
    for (int c = 0; c < 8; ++c) acc3[c] = 0.0f;

    #pragma unroll 1
    for (int d = 0; d < D_H; d += 4) {
        v4f f = *(const v4f*)(feat + d);
        #pragma unroll
        for (int c = 0; c < 8; ++c) {
            v4f w = *(const v4f*)(wbase + c * D_H + d);
            acc3[c] = fmaf(f.x, w.x, acc3[c]);
            acc3[c] = fmaf(f.y, w.y, acc3[c]);
            acc3[c] = fmaf(f.z, w.z, acc3[c]);
            acc3[c] = fmaf(f.w, w.w, acc3[c]);
        }
    }

    const float* bhp = bh + (size_t)hidx * NC + h * 8;
    float* op = out + (size_t)row * NC + h * 8;
    v4f o0 = { acc3[0] + bhp[0], acc3[1] + bhp[1], acc3[2] + bhp[2], acc3[3] + bhp[3] };
    v4f o1 = { acc3[4] + bhp[4], acc3[5] + bhp[5], acc3[6] + bhp[6], acc3[7] + bhp[7] };
    *(v4f*)(op + 0) = o0;
    *(v4f*)(op + 4) = o1;
}

extern "C" void kernel_launch(void* const* d_in, const int* in_sizes, int n_in,
                              void* d_out, int out_size, void* d_ws, size_t ws_size,
                              hipStream_t stream) {
    const float* x   = (const float*)d_in[0];
    const int*   qid = (const int*)  d_in[1];
    const int*   isc = (const int*)  d_in[2];
    const float* W1  = (const float*)d_in[3];
    const float* b1  = (const float*)d_in[4];
    const float* W2  = (const float*)d_in[5];
    const float* b2  = (const float*)d_in[6];
    const float* Wh  = (const float*)d_in[7];
    const float* bh  = (const float*)d_in[8];
    float* out = (float*)d_out;

    // workspace layout (u32): [W1H | W1L | W2H | W2L]  (~456 KB)
    uint32_t* ws  = (uint32_t*)d_ws;
    uint32_t* w1h = ws;
    uint32_t* w1l = ws + W1_PACK_N;
    uint32_t* w2h = ws + 2 * W1_PACK_N;
    uint32_t* w2l = ws + 2 * W1_PACK_N + W2_PACK_N;

    // 1) pack weights into WMMA f16 hi/lo fragment layout (deterministic, every call)
    hipLaunchKernelGGL(pack_w_f16_kernel, dim3((W1_PACK_N + 255) / 256), dim3(256),
                       0, stream, W1, D_IN, W1_PACK_N, w1h, w1l);
    hipLaunchKernelGGL(pack_w_f16_kernel, dim3((W2_PACK_N + 255) / 256), dim3(256),
                       0, stream, W2, D_H, W2_PACK_N, w2h, w2l);

    // 2) fused MLP + routed heads
    const int B = in_sizes[1];                                   // question_ids length
    const int rows_per_block = WAVES_PER_BLOCK * ROWS_PER_WAVE;  // 64
    hipLaunchKernelGGL(qmlp_fused_kernel, dim3((unsigned)(B / rows_per_block)), dim3(128),
                       0, stream,
                       x, qid, isc, w1h, w1l, b1, w2h, w2l, b2, Wh, bh, out);
}
